// MultiHeadAttention_77747497992754
// MI455X (gfx1250) — compile-verified, matrix operand order checked by reading, not run on hardware
//
#include <hip/hip_runtime.h>

#define S_LEN 2048
#define D_DIM 1024
#define NHEAD 16
#define HDIM  64
#define BTOT  2
#define MROWS (BTOT * S_LEN) // 4096

typedef __bf16 bf16_t;
typedef __attribute__((ext_vector_type(16))) __bf16 v16bf;
typedef __attribute__((ext_vector_type(8)))  float  v8f;
typedef __attribute__((ext_vector_type(4)))  unsigned int v4u;
typedef __attribute__((ext_vector_type(8)))  int v8i;
typedef __attribute__((ext_vector_type(4)))  int v4i;

#if defined(__has_builtin)
#if __has_builtin(__builtin_amdgcn_tensor_load_to_lds)
#define USE_TDM 1
#endif
#if __has_builtin(__builtin_amdgcn_permlane16)
#define USE_PERMLANE 1
#endif
#endif

union Frag {
    v16bf v;
    uint4 q[2];
};

// 16-element bf16 fragment: elements 0..7 from p[0..7], 8..15 from p[16..23]
// (CDNA5 16-bit A/B matrix VGPR layout). p must be 16B aligned.
__device__ __forceinline__ v16bf load_frag16(const bf16_t* p) {
    Frag f;
    f.q[0] = *reinterpret_cast<const uint4*>(p);
    f.q[1] = *reinterpret_cast<const uint4*>(p + 16);
    return f.v;
}

__device__ __forceinline__ v8f wmma_bf16(v16bf a, v16bf b, v8f c) {
    return __builtin_amdgcn_wmma_f32_16x16x32_bf16(false, a, false, b, (short)0, c,
                                                   false, false);
}

// ---------------------------------------------------------------------------
// 16-lane butterfly reductions. v_permlane16_b32 keeps these on the VALU pipe
// (co-executes with XDL WMMA) instead of ds_bpermute on the LDS pipe.
// ---------------------------------------------------------------------------
#ifdef USE_PERMLANE
__device__ __forceinline__ float perm16(float x, unsigned lo, unsigned hi) {
    unsigned u = __float_as_uint(x);
    u = __builtin_amdgcn_permlane16(u, u, lo, hi, false, false);
    return __uint_as_float(u);
}
__device__ __forceinline__ float row_max16(float v) {
    v = fmaxf(v, perm16(v, 0x67452301u, 0xEFCDAB89u)); // xor 1
    v = fmaxf(v, perm16(v, 0x54761032u, 0xDCFE98BAu)); // xor 2
    v = fmaxf(v, perm16(v, 0x32107654u, 0xBA98FEDCu)); // xor 4
    v = fmaxf(v, perm16(v, 0xFEDCBA98u, 0x76543210u)); // xor 8
    return v;
}
__device__ __forceinline__ float row_sum16(float v) {
    v += perm16(v, 0x67452301u, 0xEFCDAB89u);
    v += perm16(v, 0x54761032u, 0xDCFE98BAu);
    v += perm16(v, 0x32107654u, 0xBA98FEDCu);
    v += perm16(v, 0xFEDCBA98u, 0x76543210u);
    return v;
}
#else
__device__ __forceinline__ float row_max16(float v) {
    v = fmaxf(v, __shfl_xor(v, 1, 32));
    v = fmaxf(v, __shfl_xor(v, 2, 32));
    v = fmaxf(v, __shfl_xor(v, 4, 32));
    v = fmaxf(v, __shfl_xor(v, 8, 32));
    return v;
}
__device__ __forceinline__ float row_sum16(float v) {
    v += __shfl_xor(v, 1, 32);
    v += __shfl_xor(v, 2, 32);
    v += __shfl_xor(v, 4, 32);
    v += __shfl_xor(v, 8, 32);
    return v;
}
#endif

// ---------------------------------------------------------------------------
// TDM descriptor builders (CDNA5 ISA ch.8). Group0: count=1, lds_addr,
// global tile addr, type=2. Group1: data_size=2B, LDS pad 4 DWORDs every
// 32 DWORDs (row stride 64 -> 72 bf16), 2-D tensor/tile dims, 48-bit stride.
// ---------------------------------------------------------------------------
#ifdef USE_TDM
__device__ __forceinline__ v4u tdm_g0(unsigned lds_addr, const void* gaddr) {
    unsigned long long ga = (unsigned long long)gaddr;
    v4u g;
    g[0] = 1u; // count=1, user descriptor
    g[1] = lds_addr;
    g[2] = (unsigned)ga;
    g[3] = (unsigned)((ga >> 32) & 0x01FFFFFFull) | 0x80000000u; // type=2
    return g;
}
__device__ __forceinline__ v8i tdm_g1(unsigned td0, unsigned td1, unsigned t0,
                                      unsigned t1, unsigned long long s0) {
    v8i g;
    // data_size=2B (code 1), pad_enable, pad_interval=32DW (code 4),
    // pad_amount=4DW (code 3); workgroup_mask=0 (not in a cluster).
    g[0] = (int)((1u << 16) | (1u << 20) | (4u << 22) | (3u << 25));
    g[1] = (int)((td0 & 0xFFFFu) << 16);                    // tensor_dim0[15:0]
    g[2] = (int)((td0 >> 16) | ((td1 & 0xFFFFu) << 16));    // dim0 hi | dim1 lo
    g[3] = (int)((td1 >> 16) | ((t0 & 0xFFFFu) << 16));     // dim1 hi | tile0
    g[4] = (int)(t1 & 0xFFFFu);                             // tile1 | tile2=0
    g[5] = (int)(s0 & 0xFFFFFFFFull);                       // dim0_stride lo32
    g[6] = (int)((s0 >> 32) & 0xFFFFull);                   // dim0_stride hi16
    g[7] = 0;                                               // dim1_stride=0
    return g;
}
#endif

// ---------------------------------------------------------------------------
// fp32 -> bf16 conversion
// ---------------------------------------------------------------------------
__global__ __launch_bounds__(256) void cvt_f32_bf16(const float* __restrict__ in,
                                                    bf16_t* __restrict__ out, int n) {
    int i = blockIdx.x * blockDim.x + threadIdx.x;
    int stride = gridDim.x * blockDim.x;
    for (; i < n; i += stride) out[i] = (bf16_t)in[i];
}

// ---------------------------------------------------------------------------
// GEMM: Out = A(bf16 [M x 1024]) * W(bf16 [1024 x 1024] row-major)^T + bias.
// MODE 0: bf16 head-major [B,H,S,HD]  (Q,K); MODE 1: bf16 [B,H,HD,S] (V^T);
// MODE 2: f32 row-major [M,1024] (final projection).
// 256 threads = 8 waves; 128x128 tile; K staged 64 at a time into LDS,
// double-buffered via the Tensor Data Mover (wave 0 issues, all consume).
// ---------------------------------------------------------------------------
template <int MODE>
__global__ __launch_bounds__(256) void gemm_bf16_wmma(const bf16_t* __restrict__ A,
                                                      const bf16_t* __restrict__ Bt,
                                                      const float* __restrict__ bias,
                                                      void* __restrict__ outp) {
    constexpr int TK  = 64;
    constexpr int LDT = 72; // 64 + 8 pad bf16 per row (TDM pad / manual pad)
    constexpr int NT  = D_DIM / TK;
    __shared__ __align__(16) bf16_t As[2][128 * LDT];
    __shared__ __align__(16) bf16_t Bs[2][128 * LDT];

    const int tid  = threadIdx.x;
    const int lane = tid & 31;
    const int wid  = tid >> 5;
    const int wm   = wid & 3;
    const int wn   = wid >> 2;
    const int m0   = blockIdx.y * 128;
    const int n0   = blockIdx.x * 128;
    const int lrow = lane & 15;
    const int hi8  = (lane >> 4) << 3;

    v8f zero = {0.f, 0.f, 0.f, 0.f, 0.f, 0.f, 0.f, 0.f};
    v8f acc[2][4];
#pragma unroll
    for (int i = 0; i < 2; ++i)
#pragma unroll
        for (int j = 0; j < 4; ++j) acc[i][j] = zero;

#ifdef USE_TDM
    const v8i g1 = tdm_g1(/*tensor_dim0=*/D_DIM, /*tensor_dim1=*/128,
                          /*tile_dim0=*/TK, /*tile_dim1=*/128,
                          /*dim0_stride=*/D_DIM);
    const v4i z4 = {0, 0, 0, 0};
    const v8i z8 = {0, 0, 0, 0, 0, 0, 0, 0};
    auto issue_tile = [&](int t, int buf) {
        unsigned ldsA = (unsigned)(unsigned long long)(const void*)&As[buf][0];
        unsigned ldsB = (unsigned)(unsigned long long)(const void*)&Bs[buf][0];
        __builtin_amdgcn_tensor_load_to_lds(
            tdm_g0(ldsA, &A[(size_t)m0 * D_DIM + t * TK]), g1, z4, z4, z8, 0);
        __builtin_amdgcn_tensor_load_to_lds(
            tdm_g0(ldsB, &Bt[(size_t)n0 * D_DIM + t * TK]), g1, z4, z4, z8, 0);
    };
    if (wid == 0) issue_tile(0, 0);
#endif

    for (int t = 0; t < NT; ++t) {
        const int cur = t & 1;
#ifdef USE_TDM
        if (wid == 0) {
            if (t + 1 < NT) {
                issue_tile(t + 1, cur ^ 1);
                __builtin_amdgcn_s_wait_tensorcnt(2); // tile t's two loads done
            } else {
                __builtin_amdgcn_s_wait_tensorcnt(0);
            }
        }
        __syncthreads();
#else
        // Fallback: synchronous VMEM->VGPR->LDS staging.
#pragma unroll
        for (int c = 0; c < 2; ++c) {
            int chunk = tid + c * 256;
            int row   = chunk >> 2;
            int qoff  = (chunk & 3) * 16;
            *reinterpret_cast<uint4*>(&As[cur][row * LDT + qoff]) =
                *reinterpret_cast<const uint4*>(
                    &A[(size_t)(m0 + row) * D_DIM + t * TK + qoff]);
            *reinterpret_cast<uint4*>(&Bs[cur][row * LDT + qoff]) =
                *reinterpret_cast<const uint4*>(
                    &Bt[(size_t)(n0 + row) * D_DIM + t * TK + qoff]);
        }
        __syncthreads();
#endif

#pragma unroll
        for (int kt = 0; kt < 2; ++kt) {
            const bf16_t* ap = &As[cur][(wm * 32 + lrow) * LDT + kt * 32 + hi8];
            v16bf a0 = load_frag16(ap);
            v16bf a1 = load_frag16(ap + 16 * LDT);
#pragma unroll
            for (int ct = 0; ct < 4; ++ct) {
                v16bf b =
                    load_frag16(&Bs[cur][(wn * 64 + ct * 16 + lrow) * LDT + kt * 32 + hi8]);
                acc[0][ct] = wmma_bf16(a0, b, acc[0][ct]);
                acc[1][ct] = wmma_bf16(a1, b, acc[1][ct]);
            }
        }
        __syncthreads();
    }

    // Epilogue: D element r -> row (r + 8*(lane>>4)), col (lane&15).
#pragma unroll
    for (int rt = 0; rt < 2; ++rt) {
#pragma unroll
        for (int ct = 0; ct < 4; ++ct) {
            const int gn  = n0 + wn * 64 + ct * 16 + lrow;
            const float bn = bias[gn];
#pragma unroll
            for (int r = 0; r < 8; ++r) {
                const int gm = m0 + wm * 32 + rt * 16 + r + hi8;
                float v = acc[rt][ct][r] + bn;
                if (MODE == 2) {
                    ((float*)outp)[(size_t)gm * D_DIM + gn] = v;
                } else {
                    const int b = gm >> 11;
                    const int s = gm & (S_LEN - 1);
                    const int h = gn >> 6;
                    const int d = gn & (HDIM - 1);
                    bf16_t* o = (bf16_t*)outp;
                    if (MODE == 0)
                        o[(((size_t)(b * NHEAD + h)) * S_LEN + s) * HDIM + d] = (bf16_t)v;
                    else
                        o[(((size_t)(b * NHEAD + h)) * HDIM + d) * S_LEN + s] = (bf16_t)v;
                }
            }
        }
    }
}

// ---------------------------------------------------------------------------
// Flash attention: one wave owns 16 query rows; key tiles of 32; online softmax.
// Q,K: bf16 [B,H,S,64]; Vt: bf16 [B,H,64,S]; Aout: bf16 [B,S,1024].
// ---------------------------------------------------------------------------
__global__ __launch_bounds__(256) void attn_fwd(const bf16_t* __restrict__ Q,
                                                const bf16_t* __restrict__ K,
                                                const bf16_t* __restrict__ Vt,
                                                bf16_t* __restrict__ Aout) {
    __shared__ __align__(16) bf16_t Plds[8][16 * 40];

    const int tid  = threadIdx.x;
    const int lane = tid & 31;
    const int wid  = tid >> 5;
    const int lrow = lane & 15;
    const int hi8  = (lane >> 4) << 3;
    const int bh   = blockIdx.y;
    const int q0   = blockIdx.x * 128 + wid * 16;

    const bf16_t* Qb = Q + ((size_t)bh * S_LEN + q0) * HDIM;
    const bf16_t* Kb = K + (size_t)bh * S_LEN * HDIM;
    const bf16_t* Vb = Vt + (size_t)bh * HDIM * S_LEN;

    const v16bf qa0 = load_frag16(Qb + lrow * HDIM + hi8);
    const v16bf qa1 = load_frag16(Qb + lrow * HDIM + 32 + hi8);

    v8f zero = {0.f, 0.f, 0.f, 0.f, 0.f, 0.f, 0.f, 0.f};
    v8f o[4];
#pragma unroll
    for (int n = 0; n < 4; ++n) o[n] = zero;

    float m_run[8], l_run[8];
#pragma unroll
    for (int r = 0; r < 8; ++r) { m_run[r] = -3.0e38f; l_run[r] = 0.f; }
    const float scale = 0.125f; // 1/sqrt(64)

    bf16_t* Pw = &Plds[wid][0];

    for (int j = 0; j < S_LEN / 32; ++j) {
        const bf16_t* Kt = Kb + (size_t)(j * 32) * HDIM;
        v8f s0 = zero, s1 = zero;
        s0 = wmma_bf16(qa0, load_frag16(Kt + lrow * HDIM + hi8), s0);
        s0 = wmma_bf16(qa1, load_frag16(Kt + lrow * HDIM + 32 + hi8), s0);
        s1 = wmma_bf16(qa0, load_frag16(Kt + (16 + lrow) * HDIM + hi8), s1);
        s1 = wmma_bf16(qa1, load_frag16(Kt + (16 + lrow) * HDIM + 32 + hi8), s1);

        float corr[8];
#pragma unroll
        for (int r = 0; r < 8; ++r) {
            float a = s0[r] * scale;
            float b = s1[r] * scale;
            float mx = row_max16(fmaxf(a, b));
            float mn = fmaxf(m_run[r], mx);
            corr[r]  = __expf(m_run[r] - mn);
            m_run[r] = mn;
            float p0 = __expf(a - mn);
            float p1 = __expf(b - mn);
            float rs = row_sum16(p0 + p1);
            l_run[r] = l_run[r] * corr[r] + rs;
            const int row = r + hi8;
            Pw[row * 40 + lrow]      = (bf16_t)p0;
            Pw[row * 40 + 16 + lrow] = (bf16_t)p1;
        }
#pragma unroll
        for (int n = 0; n < 4; ++n)
#pragma unroll
            for (int r = 0; r < 8; ++r) o[n][r] *= corr[r];

        // Same-wave LDS store->load ordering; fence compiler + DS counter.
        __asm__ volatile("s_wait_dscnt 0" ::: "memory");

        v16bf pa = load_frag16(Pw + lrow * 40 + hi8);
#pragma unroll
        for (int n = 0; n < 4; ++n) {
            v16bf vb = load_frag16(Vb + (size_t)(n * 16 + lrow) * S_LEN + j * 32 + hi8);
            o[n] = wmma_bf16(pa, vb, o[n]);
        }
    }

    const int b = bh >> 4;
    const int h = bh & 15;
#pragma unroll
    for (int r = 0; r < 8; ++r) {
        const float inv = 1.0f / l_run[r];
        const int s = q0 + r + hi8;
        bf16_t* op = Aout + ((size_t)(b * S_LEN + s)) * D_DIM + h * HDIM;
#pragma unroll
        for (int n = 0; n < 4; ++n) op[n * 16 + lrow] = (bf16_t)(o[n][r] * inv);
    }
}

// ---------------------------------------------------------------------------
extern "C" void kernel_launch(void* const* d_in, const int* in_sizes, int n_in,
                              void* d_out, int out_size, void* d_ws, size_t ws_size,
                              hipStream_t stream) {
    const float* x  = (const float*)d_in[0];
    const float* Wq = (const float*)d_in[1];
    const float* bq = (const float*)d_in[2];
    const float* Wk = (const float*)d_in[3];
    const float* bk = (const float*)d_in[4];
    const float* Wv = (const float*)d_in[5];
    const float* bv = (const float*)d_in[6];
    const float* Wo = (const float*)d_in[7];
    const float* bo = (const float*)d_in[8];

    char* ws = (char*)d_ws;
    size_t off = 0;
    auto carve = [&](size_t bytes) {
        char* p = ws + off;
        off += (bytes + 255) & ~(size_t)255;
        return p;
    };
    bf16_t* xb  = (bf16_t*)carve((size_t)MROWS * D_DIM * 2);
    bf16_t* wqb = (bf16_t*)carve((size_t)D_DIM * D_DIM * 2);
    bf16_t* wkb = (bf16_t*)carve((size_t)D_DIM * D_DIM * 2);
    bf16_t* wvb = (bf16_t*)carve((size_t)D_DIM * D_DIM * 2);
    bf16_t* wob = (bf16_t*)carve((size_t)D_DIM * D_DIM * 2);
    bf16_t* Qh  = (bf16_t*)carve((size_t)MROWS * D_DIM * 2); // [B,H,S,64]
    bf16_t* Kh  = (bf16_t*)carve((size_t)MROWS * D_DIM * 2); // [B,H,S,64]
    bf16_t* Vt  = (bf16_t*)carve((size_t)MROWS * D_DIM * 2); // [B,H,64,S]
    bf16_t* AO  = (bf16_t*)carve((size_t)MROWS * D_DIM * 2); // [B,S,1024]

    cvt_f32_bf16<<<256, 256, 0, stream>>>(x, xb, MROWS * D_DIM);
    cvt_f32_bf16<<<64, 256, 0, stream>>>(Wq, wqb, D_DIM * D_DIM);
    cvt_f32_bf16<<<64, 256, 0, stream>>>(Wk, wkb, D_DIM * D_DIM);
    cvt_f32_bf16<<<64, 256, 0, stream>>>(Wv, wvb, D_DIM * D_DIM);
    cvt_f32_bf16<<<64, 256, 0, stream>>>(Wo, wob, D_DIM * D_DIM);

    dim3 gg(D_DIM / 128, MROWS / 128); // (8, 32)
    gemm_bf16_wmma<0><<<gg, 256, 0, stream>>>(xb, wqb, bq, Qh);
    gemm_bf16_wmma<0><<<gg, 256, 0, stream>>>(xb, wkb, bk, Kh);
    gemm_bf16_wmma<1><<<gg, 256, 0, stream>>>(xb, wvb, bv, Vt);

    attn_fwd<<<dim3(S_LEN / 128, BTOT * NHEAD), 256, 0, stream>>>(Qh, Kh, Vt, AO);

    gemm_bf16_wmma<2><<<gg, 256, 0, stream>>>(AO, wob, bo, d_out);
}
